// MoELayer_47914655154654
// MI455X (gfx1250) — matrix operand order repaired and checked by reading, not compiled, not run on hardware
//
#include <hip/hip_runtime.h>
#include <hip/hip_bf16.h>

// ---------------------------------------------------------------------------
// Hierarchical MoE (2 levels), top-2 routing, D=512, DFF=2048, E=(4,8).
// Exact algebra of the reference:
//   out[t] = sum_e p[t,e]*c_e + sum_{e in top2(t)} p[t,e]*(FFN_e(x_t) - c_e)
// with c_e = relu(b1_e) @ W2_e + b2_e (expert output on masked/zero input).
// Routed FFN uses v_wmma_f32_16x16x32_bf16 (fp32 accumulate). Weights are
// pre-transposed+converted to bf16 so B fragments load as two b128 directly
// from global (L2-resident), in contiguous k order (k = 16*half + s), same as the A fragments.
// ---------------------------------------------------------------------------

#define T_TOK 8192
#define DM    512
#define DFFM  2048
#define E0C   4
#define E1C   8
#define FCH   128   // DFF chunk width
#define MT    48    // token tile (three 16-row WMMA M tiles)

typedef __attribute__((ext_vector_type(16))) __bf16 v16bf;
typedef __attribute__((ext_vector_type(8)))  float  v8f;

union Frag16 { unsigned int u[8]; uint4 q[2]; v16bf v; };

__device__ __forceinline__ unsigned short f2bf(float f) {
  __bf16 b = (__bf16)f;                       // RNE convert on gfx1250
  return __builtin_bit_cast(unsigned short, b);
}

// ---------------------------------------------------------------------------
__global__ void cvt_f32_bf16_kernel(const float* __restrict__ in,
                                    unsigned short* __restrict__ out, int n) {
  int i = blockIdx.x * blockDim.x + threadIdx.x;
  if (i < n) out[i] = f2bf(in[i]);
}

// Tiled transpose + f32->bf16: in[e][R][C] -> out[e][C][R]. Grid: (C/32,R/32,E).
__global__ void transpose_cvt_kernel(const float* __restrict__ in,
                                     unsigned short* __restrict__ out,
                                     int R, int C) {
  __shared__ float tile[32][33];
  int e = blockIdx.z;
  const float* inp = in + (size_t)e * R * C;
  unsigned short* outp = out + (size_t)e * R * C;
  int c0 = blockIdx.x * 32, r0 = blockIdx.y * 32;
  int tx = threadIdx.x, ty = threadIdx.y;     // 32 x 8
#pragma unroll
  for (int i = 0; i < 32; i += 8)
    tile[ty + i][tx] = inp[(size_t)(r0 + ty + i) * C + (c0 + tx)];
  __syncthreads();
#pragma unroll
  for (int i = 0; i < 32; i += 8)
    outp[(size_t)(c0 + ty + i) * R + (r0 + tx)] = f2bf(tile[tx][ty + i]);
}

// c_e[d] = b2[e,d] + sum_f relu(b1[e,f]) * W2[e,f,d]   (exact fp32)
__global__ void expert_const_kernel(const float* __restrict__ b1,
                                    const float* __restrict__ W2,
                                    const float* __restrict__ b2,
                                    float* __restrict__ c) {
  int e = blockIdx.y;
  int d = blockIdx.x * blockDim.x + threadIdx.x;       // 0..511
  float acc = b2[(size_t)e * DM + d];
  const float* w = W2 + (size_t)e * DFFM * DM + d;
  const float* b = b1 + (size_t)e * DFFM;
  for (int f = 0; f < DFFM; ++f) {
    float r = b[f];
    r = r > 0.f ? r : 0.f;
    acc += r * w[(size_t)f * DM];
  }
  c[(size_t)e * DM + d] = acc;
}

// One wave32 per token: logits -> softmax -> top-2 -> probs + expert lists.
template <int E>
__global__ void router_kernel(const float* __restrict__ x,
                              const float* __restrict__ Wr,
                              float* __restrict__ probs,
                              int* __restrict__ lists,
                              int* __restrict__ counts) {
  int wave = threadIdx.x >> 5;
  int lane = threadIdx.x & 31;
  int t = blockIdx.x * (blockDim.x >> 5) + wave;
  if (t >= T_TOK) return;

  float acc[E];
#pragma unroll
  for (int e = 0; e < E; ++e) acc[e] = 0.f;
  const float* xr = x + (size_t)t * DM;
  for (int i = lane; i < DM; i += 32) {
    float xv = xr[i];
#pragma unroll
    for (int e = 0; e < E; ++e) acc[e] += xv * Wr[(size_t)i * E + e];
  }
#pragma unroll
  for (int e = 0; e < E; ++e)
    for (int off = 16; off; off >>= 1) acc[e] += __shfl_xor(acc[e], off, 32);

  if (lane == 0) {
    float mx = acc[0];
#pragma unroll
    for (int e = 1; e < E; ++e) mx = fmaxf(mx, acc[e]);
    float p[E]; float s = 0.f;
#pragma unroll
    for (int e = 0; e < E; ++e) { p[e] = __expf(acc[e] - mx); s += p[e]; }
    float inv = 1.f / s;
    int i1 = 0;
#pragma unroll
    for (int e = 1; e < E; ++e) if (p[e] > p[i1]) i1 = e;
    int i2 = -1;
#pragma unroll
    for (int e = 0; e < E; ++e) {
      if (e == i1) continue;
      if (i2 < 0 || p[e] > p[i2]) i2 = e;
    }
#pragma unroll
    for (int e = 0; e < E; ++e) probs[(size_t)t * E + e] = p[e] * inv;
    int s1 = atomicAdd(&counts[i1], 1);
    lists[(size_t)i1 * T_TOK + s1] = t;
    int s2 = atomicAdd(&counts[i2], 1);
    lists[(size_t)i2 * T_TOK + s2] = t;
  }
}

// out[t,d] = sum_e p[t,e] * c[e,d]   (dense constant part; full overwrite)
template <int E>
__global__ void base_kernel(const float* __restrict__ probs,
                            const float* __restrict__ c,
                            float* __restrict__ out) {
  int lin = blockIdx.x * blockDim.x + threadIdx.x;
  if (lin >= T_TOK * DM) return;
  int t = lin >> 9;           // /512
  int d = lin & (DM - 1);
  float acc = 0.f;
#pragma unroll
  for (int e = 0; e < E; ++e)
    acc += probs[(size_t)t * E + e] * c[(size_t)e * DM + d];
  out[lin] = acc;
}

// ---------------------------------------------------------------------------
// Routed FFN: block = (expert, 48-token tile), 256 threads = 8 wave32.
// Fused x@W1 -> relu -> @W2, DFF chunked by 128. B fragments come straight
// from pre-transposed bf16 weights as two global b128 loads per fragment.
// ---------------------------------------------------------------------------
template <int E>
__global__ void __launch_bounds__(256)
moe_ffn_kernel(const unsigned short* __restrict__ xb,
               const unsigned short* __restrict__ W1t,   // [E][DFF][D] bf16
               const float* __restrict__ b1,
               const unsigned short* __restrict__ W2t,   // [E][D][DFF] bf16
               const float* __restrict__ b2,
               const float* __restrict__ probs, const float* __restrict__ cvec,
               const int* __restrict__ lists, const int* __restrict__ counts,
               float* __restrict__ out) {
  __shared__ __align__(16) unsigned short Aq[MT][DM];    // 48 KB, x tile (bf16)
  __shared__ __align__(16) unsigned short Hq[MT][FCH];   // 12 KB, H chunk (bf16)
  __shared__ int toks[MT];

  const int e = blockIdx.y;
  const int cnt = counts[e];
  const int tile0 = blockIdx.x * MT;
  if (tile0 >= cnt) return;

  const int tid  = threadIdx.x;
  const int lane = tid & 31;
  const int w    = tid >> 5;            // wave 0..7
  const int half = (lane >> 4) & 1;
  const int l16  = lane & 15;

  if (tid < MT) {
    int idx = tile0 + tid;
    toks[tid] = (idx < cnt) ? lists[(size_t)e * T_TOK + idx] : -1;
  }
  __syncthreads();

  // Gather token rows (already bf16) into LDS A tile; zero invalid rows.
  {
    int cb = (tid & 15) * 32;           // 32 bf16 per thread per row
    for (int r = tid >> 4; r < MT; r += 16) {
      int tok = toks[r];
      unsigned int* dst = (unsigned int*)&Aq[r][cb];
      if (tok >= 0) {
        const unsigned int* src = (const unsigned int*)(xb + (size_t)tok * DM + cb);
#pragma unroll
        for (int i = 0; i < 16; ++i) dst[i] = src[i];
      } else {
#pragma unroll
        for (int i = 0; i < 16; ++i) dst[i] = 0u;
      }
    }
  }
  __syncthreads();

  const unsigned short* W1te = W1t + (size_t)e * DFFM * DM;
  const unsigned short* W2te = W2t + (size_t)e * DM * DFFM;
  const float* b1e = b1 + (size_t)e * DFFM;

  v8f acc[12];                          // [mt*4 + nt], Y tile accumulators
#pragma unroll
  for (int i = 0; i < 12; ++i) acc[i] = {};

  for (int fc = 0; fc < DFFM; fc += FCH) {
    if (fc + FCH < DFFM) {              // prefetch next chunk's weight rows
      __builtin_prefetch(W1te + (size_t)(fc + FCH + w * 16 + l16) * DM, 0, 1);
      __builtin_prefetch(W2te + (size_t)(w * 64 + l16) * DFFM + fc + FCH, 0, 1);
    }

    // ---- GEMM1: H[48 x FCH] = relu(A @ W1t^T + b1); wave w owns 16 cols ----
    v8f hacc[3] = {{}, {}, {}};
    const unsigned short* bcol1 =
        W1te + (size_t)(fc + w * 16 + l16) * DM + half * 16;
    for (int k0 = 0; k0 < DM; k0 += 32) {
      Frag16 fb;                        // B fragment: 32 contiguous bytes
      fb.q[0] = ((const uint4*)(bcol1 + k0))[0];
      fb.q[1] = ((const uint4*)(bcol1 + k0))[1];
#pragma unroll
      for (int mt = 0; mt < 3; ++mt) {
        Frag16 fa;
        const unsigned short* ap = &Aq[mt * 16 + l16][k0 + half * 16];
#pragma unroll
        for (int j = 0; j < 8; ++j) {
          int k = j * 2;
          fa.u[j] = *(const unsigned int*)(ap + k);
        }
        hacc[mt] = __builtin_amdgcn_wmma_f32_16x16x32_bf16(
            false, fa.v, false, fb.v, (short)0, hacc[mt], false, false);
      }
    }
    __syncthreads();                    // prior GEMM2 reads of Hq are done
    {
      float bb = b1e[fc + w * 16 + l16];
#pragma unroll
      for (int mt = 0; mt < 3; ++mt) {
#pragma unroll
        for (int r = 0; r < 8; ++r) {
          int m = mt * 16 + half * 8 + r;
          float v = hacc[mt][r] + bb;
          v = v > 0.f ? v : 0.f;
          Hq[m][w * 16 + l16] = f2bf(v);
        }
      }
    }
    __syncthreads();

    // ---- GEMM2: Y += H_chunk @ W2t^T; wave w owns 64 output cols ----
    for (int ks = 0; ks < FCH; ks += 32) {
      Frag16 fa[3];
#pragma unroll
      for (int mt = 0; mt < 3; ++mt) {
        const unsigned short* ap = &Hq[mt * 16 + l16][ks + half * 16];
#pragma unroll
        for (int j = 0; j < 8; ++j) {
          int k = j * 2;
          fa[mt].u[j] = *(const unsigned int*)(ap + k);
        }
      }
#pragma unroll
      for (int nt = 0; nt < 4; ++nt) {
        Frag16 fb;
        const unsigned short* bp =
            W2te + (size_t)(w * 64 + nt * 16 + l16) * DFFM + fc + ks + half * 16;
        fb.q[0] = ((const uint4*)bp)[0];
        fb.q[1] = ((const uint4*)bp)[1];
#pragma unroll
        for (int mt = 0; mt < 3; ++mt) {
          acc[mt * 4 + nt] = __builtin_amdgcn_wmma_f32_16x16x32_bf16(
              false, fa[mt].v, false, fb.v, (short)0, acc[mt * 4 + nt], false, false);
        }
      }
    }
  }

  // ---- epilogue: out[tok, n] += p * (Y + b2 - c_e) ----
#pragma unroll
  for (int nt = 0; nt < 4; ++nt) {
    int n = w * 64 + nt * 16 + l16;
    float bc = b2[(size_t)e * DM + n] - cvec[(size_t)e * DM + n];
#pragma unroll
    for (int mt = 0; mt < 3; ++mt) {
#pragma unroll
      for (int r = 0; r < 8; ++r) {
        int m = mt * 16 + half * 8 + r;
        int tok = toks[m];
        if (tok < 0) continue;
        float p = probs[(size_t)tok * E + e];
        unsafeAtomicAdd(&out[(size_t)tok * DM + n], p * (acc[mt * 4 + nt][r] + bc));
      }
    }
  }
}

// ---------------------------------------------------------------------------
extern "C" void kernel_launch(void* const* d_in, const int* in_sizes, int n_in,
                              void* d_out, int out_size, void* d_ws, size_t ws_size,
                              hipStream_t stream) {
  const float* x    = (const float*)d_in[0];
  const float* Wr0  = (const float*)d_in[1];
  const float* W1_0 = (const float*)d_in[2];
  const float* b1_0 = (const float*)d_in[3];
  const float* W2_0 = (const float*)d_in[4];
  const float* b2_0 = (const float*)d_in[5];
  const float* Wr1  = (const float*)d_in[6];
  const float* W1_1 = (const float*)d_in[7];
  const float* b1_1 = (const float*)d_in[8];
  const float* W2_1 = (const float*)d_in[9];
  const float* b2_1 = (const float*)d_in[10];
  float* out = (float*)d_out;

  char* ws = (char*)d_ws;
  size_t off = 0;
  auto alloc = [&](size_t bytes) -> void* {
    void* p = ws + off;
    off += (bytes + 255) & ~(size_t)255;
    return p;
  };
  unsigned short* xb   = (unsigned short*)alloc((size_t)T_TOK * DM * 2);
  float* h0            = (float*)alloc((size_t)T_TOK * DM * 4);
  unsigned short* h0b  = (unsigned short*)alloc((size_t)T_TOK * DM * 2);
  float* probs0        = (float*)alloc((size_t)T_TOK * E0C * 4);
  float* probs1        = (float*)alloc((size_t)T_TOK * E1C * 4);
  int* lists0          = (int*)alloc((size_t)E0C * T_TOK * 4);
  int* lists1          = (int*)alloc((size_t)E1C * T_TOK * 4);
  int* counts          = (int*)alloc((E0C + E1C) * 4);
  float* c0            = (float*)alloc((size_t)E0C * DM * 4);
  float* c1            = (float*)alloc((size_t)E1C * DM * 4);
  unsigned short* w1t0 = (unsigned short*)alloc((size_t)E0C * DM * DFFM * 2);
  unsigned short* w2t0 = (unsigned short*)alloc((size_t)E0C * DM * DFFM * 2);
  unsigned short* w1t1 = (unsigned short*)alloc((size_t)E1C * DM * DFFM * 2);
  unsigned short* w2t1 = (unsigned short*)alloc((size_t)E1C * DM * DFFM * 2);
  int* counts0 = counts;
  int* counts1 = counts + E0C;

  hipMemsetAsync(counts, 0, (E0C + E1C) * sizeof(int), stream);

  const int nElem = T_TOK * DM;
  const dim3 tb(32, 8);

  cvt_f32_bf16_kernel<<<nElem / 256, 256, 0, stream>>>(x, xb, nElem);
  // weights: [E][R][C] f32 -> [E][C][R] bf16 (B-fragment-friendly layout)
  transpose_cvt_kernel<<<dim3(DFFM / 32, DM / 32, E0C), tb, 0, stream>>>(W1_0, w1t0, DM, DFFM);
  transpose_cvt_kernel<<<dim3(DM / 32, DFFM / 32, E0C), tb, 0, stream>>>(W2_0, w2t0, DFFM, DM);
  transpose_cvt_kernel<<<dim3(DFFM / 32, DM / 32, E1C), tb, 0, stream>>>(W1_1, w1t1, DM, DFFM);
  transpose_cvt_kernel<<<dim3(DM / 32, DFFM / 32, E1C), tb, 0, stream>>>(W2_1, w2t1, DFFM, DM);
  expert_const_kernel<<<dim3(DM / 256, E0C), 256, 0, stream>>>(b1_0, W2_0, b2_0, c0);
  expert_const_kernel<<<dim3(DM / 256, E1C), 256, 0, stream>>>(b1_1, W2_1, b2_1, c1);

  const int nTiles = (T_TOK + MT - 1) / MT;

  // ----- level 0 -----
  router_kernel<E0C><<<T_TOK / 8, 256, 0, stream>>>(x, Wr0, probs0, lists0, counts0);
  base_kernel<E0C><<<nElem / 256, 256, 0, stream>>>(probs0, c0, h0);
  moe_ffn_kernel<E0C><<<dim3(nTiles, E0C), 256, 0, stream>>>(
      xb, w1t0, b1_0, w2t0, b2_0, probs0, c0, lists0, counts0, h0);

  // ----- level 1 -----
  cvt_f32_bf16_kernel<<<nElem / 256, 256, 0, stream>>>(h0, h0b, nElem);
  router_kernel<E1C><<<T_TOK / 8, 256, 0, stream>>>(h0, Wr1, probs1, lists1, counts1);
  base_kernel<E1C><<<nElem / 256, 256, 0, stream>>>(probs1, c1, out);
  moe_ffn_kernel<E1C><<<dim3(nTiles, E1C), 256, 0, stream>>>(
      h0b, w1t1, b1_1, w2t1, b2_1, probs1, c1, lists1, counts1, out);
}
